// Recurrence_64115271795052
// MI455X (gfx1250) — matrix operand order repaired and checked by reading, not compiled, not run on hardware
//
#include <hip/hip_runtime.h>

typedef __attribute__((ext_vector_type(16))) __bf16       v16bf;
typedef __attribute__((ext_vector_type(8)))  float        v8f;
typedef __attribute__((ext_vector_type(4)))  unsigned int v4u;
typedef __attribute__((ext_vector_type(4)))  unsigned int u32x4;
typedef __attribute__((ext_vector_type(8)))  int          i32x8;
typedef __attribute__((ext_vector_type(4)))  int          i32x4;

// addrspace(3) (LDS) typed views so fragment loads lower to ds_load_b128.
typedef __attribute__((address_space(3))) unsigned short lds_u16;
typedef __attribute__((address_space(3))) v4u            lds_v4u;

union FragBF { v4u u[2]; v16bf bf; };

__device__ __forceinline__ unsigned short f2bf(float f) {
  unsigned int u = __float_as_uint(f);
  u += 0x7FFFu + ((u >> 16) & 1u);          // round-to-nearest-even
  return (unsigned short)(u >> 16);
}

// LDS byte offset of a __shared__ object (addrspace(3) offset).
__device__ __forceinline__ unsigned lds_off_of(const void* p) {
  return (unsigned)(unsigned long long)(const __attribute__((address_space(3))) void*)p;
}

#if __has_builtin(__builtin_amdgcn_tensor_load_to_lds) && \
    __has_builtin(__builtin_amdgcn_s_wait_tensorcnt)
#define HAVE_TDM 1
#else
#define HAVE_TDM 0
#endif

#if HAVE_TDM
// Issue one 2D TDM tile load: tile_d1 rows x tile_d0 elems (2-byte elems),
// row stride d0_stride elems in memory, contiguous rows in LDS with hardware
// padding of pad_amount(code) DWORDs every pad_interval(code) DWORDs.
__device__ __forceinline__ void tdm_load_2d(
    unsigned lds_byte, unsigned long long gaddr,
    unsigned tensor_d0, unsigned tensor_d1, unsigned long long d0_stride,
    unsigned tile_d0, unsigned tile_d1,
    unsigned pad_interval_code, unsigned pad_amount_code) {
  u32x4 g0;
  g0[0] = 1u;                                       // count=1, user mode
  g0[1] = lds_byte;                                 // lds_addr
  g0[2] = (unsigned)gaddr;                          // global_addr[31:0]
  g0[3] = (unsigned)((gaddr >> 32) & 0x01FFFFFFull) // global_addr[56:32]
          | 0x80000000u;                            // type=2 ("image")
  i32x8 g1;
  g1[0] = (int)((1u << 16)                          // data_size = 2 bytes
                | (1u << 20)                        // pad_enable
                | (pad_interval_code << 22) | (pad_amount_code << 25));
  g1[1] = (int)((tensor_d0 & 0xFFFFu) << 16);       // tensor_dim0 lo16 @48
  g1[2] = (int)(((tensor_d0 >> 16) & 0xFFFFu) | ((tensor_d1 & 0xFFFFu) << 16));
  g1[3] = (int)(((tensor_d1 >> 16) & 0xFFFFu) | (tile_d0 << 16));
  g1[4] = (int)tile_d1;                             // tile_dim1 (tile_dim2=0)
  g1[5] = (int)(unsigned)d0_stride;                 // tensor_dim0_stride lo32
  g1[6] = (int)((unsigned)(d0_stride >> 32) & 0xFFFFu);
  g1[7] = 0;
  i32x4 z4 = {0, 0, 0, 0};
#if __has_include(<hip/amd_detail/amd_gfx1250_TDM.h>)
  i32x8 z8 = {0, 0, 0, 0, 0, 0, 0, 0};              // therock 6-arg form
  __builtin_amdgcn_tensor_load_to_lds(g0, g1, z4, z4, z8, 0);
#else
  __builtin_amdgcn_tensor_load_to_lds(g0, g1, z4, z4, 0); // ROCm 7.2 5-arg form
#endif
}
#endif  // HAVE_TDM

// ---------------------------------------------------------------------------
// Kernel 1: P = softmax(T_param, axis=1) -> bf16 [1024, 4096]; keep rowmax /
// rowsum (f32) so the argmax step can rebuild P exactly.
// ---------------------------------------------------------------------------
__global__ __launch_bounds__(256) void softmax_p_kernel(
    const float* __restrict__ T, unsigned short* __restrict__ P,
    float* __restrict__ rowmax, float* __restrict__ rowsum) {
  __shared__ float red[4][256];
  const int k = blockIdx.x, t = threadIdx.x;
  float4 v[4];
  float mx[4] = {-3.4e38f, -3.4e38f, -3.4e38f, -3.4e38f};
#pragma unroll
  for (int si = 0; si < 4; ++si) {
    int s = t + si * 256;
    v[si] = *(const float4*)(T + (size_t)k * 4096 + s * 4);
    mx[0] = fmaxf(mx[0], v[si].x); mx[1] = fmaxf(mx[1], v[si].y);
    mx[2] = fmaxf(mx[2], v[si].z); mx[3] = fmaxf(mx[3], v[si].w);
  }
#pragma unroll
  for (int a = 0; a < 4; ++a) red[a][t] = mx[a];
  __syncthreads();
  for (int st = 128; st > 0; st >>= 1) {
    if (t < st)
#pragma unroll
      for (int a = 0; a < 4; ++a) red[a][t] = fmaxf(red[a][t], red[a][t + st]);
    __syncthreads();
  }
  const float m0 = red[0][0], m1 = red[1][0], m2 = red[2][0], m3 = red[3][0];
  __syncthreads();
  float e[4][4];
  float s0 = 0.f, s1 = 0.f, s2 = 0.f, s3 = 0.f;
#pragma unroll
  for (int si = 0; si < 4; ++si) {
    e[si][0] = __expf(v[si].x - m0); s0 += e[si][0];
    e[si][1] = __expf(v[si].y - m1); s1 += e[si][1];
    e[si][2] = __expf(v[si].z - m2); s2 += e[si][2];
    e[si][3] = __expf(v[si].w - m3); s3 += e[si][3];
  }
  red[0][t] = s0; red[1][t] = s1; red[2][t] = s2; red[3][t] = s3;
  __syncthreads();
  for (int st = 128; st > 0; st >>= 1) {
    if (t < st)
#pragma unroll
      for (int a = 0; a < 4; ++a) red[a][t] += red[a][t + st];
    __syncthreads();
  }
  const float i0 = 1.f / red[0][0], i1 = 1.f / red[1][0];
  const float i2 = 1.f / red[2][0], i3 = 1.f / red[3][0];
#pragma unroll
  for (int si = 0; si < 4; ++si) {
    int s = t + si * 256;
    ushort4 o;
    o.x = f2bf(e[si][0] * i0); o.y = f2bf(e[si][1] * i1);
    o.z = f2bf(e[si][2] * i2); o.w = f2bf(e[si][3] * i3);
    *(ushort4*)(P + (size_t)k * 4096 + s * 4) = o;
  }
  if (t == 0) {
    rowmax[k * 4 + 0] = m0; rowmax[k * 4 + 1] = m1;
    rowmax[k * 4 + 2] = m2; rowmax[k * 4 + 3] = m3;
    rowsum[k * 4 + 0] = red[0][0]; rowsum[k * 4 + 1] = red[1][0];
    rowsum[k * 4 + 2] = red[2][0]; rowsum[k * 4 + 3] = red[3][0];
  }
}

// ---------------------------------------------------------------------------
// Kernel 2: Bellman GEMM. new_val[m,s] = rewards[m,s] + max_a (A @ P)[m,s*4+a]
// A: [2048,1024] bf16, P: [1024,4096] bf16.  BM=64 BN=128 KC=64, 8 waves/WG,
// wave = 16x64 strip (4 accumulators).  Tiles staged via TDM tensor_load_to_lds
// (double buffered on TENSORcnt, hw-padded rows); fragments prefetched through
// addrspace(3) pointers so the 8 v_wmma per chunk get ds_load_b128 feeds.
// ---------------------------------------------------------------------------
#define BGK_AS 72    // A tile LDS row stride (64 + 8 pad) in bf16 elems
#define BGK_BS 136   // B tile LDS row stride (128 + 8 pad)

__global__ __launch_bounds__(256) void bellman_gemm_kernel(
    const unsigned short* __restrict__ A, const unsigned short* __restrict__ B,
    const float* __restrict__ rewards, unsigned short* __restrict__ outB,
    float* __restrict__ outF) {
  __shared__ unsigned short lA[2][64 * BGK_AS];
  __shared__ unsigned short lB[2][64 * BGK_BS];
  const int t = threadIdx.x;
  const int lane = t & 31;
  const int w = t >> 5, wm = w & 3, wn = w >> 2;
  const int m0 = blockIdx.y * 64;
  const int n0 = blockIdx.x * 128;

  // Launder the 32-bit LDS offsets (TDM writes LDS behind the compiler's
  // back; untouched they'd be "loads of never-stored memory").  Keeping the
  // laundering in addrspace(3) preserves ds_load codegen.
  unsigned offA = lds_off_of(&lA[0][0]);
  unsigned offB = lds_off_of(&lB[0][0]);
  asm volatile("" : "+v"(offA), "+v"(offB));
  const lds_u16* laBase = (const lds_u16*)offA;
  const lds_u16* lbBase = (const lds_u16*)offB;

  v8f acc[4];
#pragma unroll
  for (int j = 0; j < 4; ++j) acc[j] = {};

  const bool issuer = (w == 0);

#if HAVE_TDM
  const unsigned ldsA[2] = {lds_off_of(&lA[0][0]), lds_off_of(&lA[1][0])};
  const unsigned ldsB[2] = {lds_off_of(&lB[0][0]), lds_off_of(&lB[1][0])};
  if (issuer) {  // prologue: chunk 0 into buffer 0
    tdm_load_2d(ldsA[0], (unsigned long long)(A + (size_t)m0 * 1024),
                1024, 2048, 1024, 64, 64, /*pad every 32 dw*/ 4, /*by 4 dw*/ 3);
    tdm_load_2d(ldsB[0], (unsigned long long)(B + n0),
                4096, 1024, 4096, 128, 64, /*pad every 64 dw*/ 5, /*by 4 dw*/ 3);
  }
#endif

  for (int k0 = 0; k0 < 1024; k0 += 64) {
    const int buf = (k0 >> 6) & 1;
#if HAVE_TDM
    // SALU wait, EXEC-independent; non-issuer waves have TENSORcnt==0.
    __builtin_amdgcn_s_wait_tensorcnt(0);
    __syncthreads();  // everyone may read buf; buf^1 free for the next TDM
    if (issuer && (k0 + 64) < 1024) {
      tdm_load_2d(ldsA[buf ^ 1],
                  (unsigned long long)(A + (size_t)m0 * 1024 + (k0 + 64)),
                  1024, 2048, 1024, 64, 64, 4, 3);
      tdm_load_2d(ldsB[buf ^ 1],
                  (unsigned long long)(B + (size_t)(k0 + 64) * 4096 + n0),
                  4096, 1024, 4096, 128, 64, 5, 3);
    }
#else
    __syncthreads();
#pragma unroll
    for (int p = 0; p < 2; ++p) {            // A tile 64x64 bf16
      int idx = t + p * 256, r = idx >> 3, c = (idx & 7) * 8;
      *(v4u*)(&lA[0][0] + r * BGK_AS + c) =
          *(const v4u*)(A + (size_t)(m0 + r) * 1024 + k0 + c);
    }
#pragma unroll
    for (int p = 0; p < 4; ++p) {            // B tile 64x128 bf16
      int idx = t + p * 256, r = idx >> 4, c = (idx & 15) * 8;
      *(v4u*)(&lB[0][0] + r * BGK_BS + c) =
          *(const v4u*)(B + (size_t)(k0 + r) * 4096 + n0 + c);
    }
    __syncthreads();
#endif
    const lds_u16* la = laBase + (HAVE_TDM ? buf : 0) * (64 * BGK_AS);
    const lds_u16* lb = lbBase + (HAVE_TDM ? buf : 0) * (64 * BGK_BS);

    // Prefetch all fragments for this chunk, then run the WMMAs back-to-back.
    FragBF af[2], bfr[2][4];
    const int arow = wm * 16 + (lane & 15);
    const int bkb = (lane >> 4) << 3;
#pragma unroll
    for (int ks = 0; ks < 2; ++ks) {
      const int ak = ks * 32 + ((lane >> 4) << 3);
      af[ks].u[0] = *(const lds_v4u*)(la + arow * BGK_AS + ak);
      af[ks].u[1] = *(const lds_v4u*)(la + arow * BGK_AS + ak + 16);
      const int bk = ks * 32 + bkb;
#pragma unroll
      for (int j = 0; j < 4; ++j) {
        const int bc = wn * 64 + j * 16 + (lane & 15);
#pragma unroll
        for (int s = 0; s < 16; ++s)
          ((unsigned short*)&bfr[ks][j])[s] = lb[(bk + (s < 8 ? s : s + 8)) * BGK_BS + bc];
      }
    }
#pragma unroll
    for (int ks = 0; ks < 2; ++ks)
#pragma unroll
      for (int j = 0; j < 4; ++j)
        acc[j] = __builtin_amdgcn_wmma_f32_16x16x32_bf16(
            false, af[ks].bf, false, bfr[ks][j].bf, (short)0, acc[j], false, false);
  }

  // epilogue: max over groups of 4 consecutive columns (na=4), + rewards
#pragma unroll
  for (int j = 0; j < 4; ++j) {
    float mx[8];
#pragma unroll
    for (int r = 0; r < 8; ++r) mx[r] = acc[j][r];
#pragma unroll
    for (int d = 1; d < 4; d <<= 1)
#pragma unroll
      for (int r = 0; r < 8; ++r) mx[r] = fmaxf(mx[r], __shfl_xor(mx[r], d, 32));
    if ((lane & 3) == 0) {
      const int s = (n0 + wn * 64 + j * 16 + (lane & 15)) >> 2;
      const int mrow = m0 + wm * 16 + ((lane >> 4) << 3);
#pragma unroll
      for (int r = 0; r < 8; ++r) {
        const size_t o = (size_t)(mrow + r) * 1024 + s;
        const float val = rewards[o] + mx[r];
        outB[o] = f2bf(val);
        outF[o] = val;
      }
    }
  }
}

// ---------------------------------------------------------------------------
// Kernel 3: MLP GEMM, out = relu(A[2048,K] @ B[K,N] + bias), bf16 in, bf16 out
// (+ optional f32 copy).  Load/ds_store staging, prefetched fragments.
// ---------------------------------------------------------------------------
__global__ __launch_bounds__(256) void mlp_gemm_kernel(
    const unsigned short* __restrict__ A, const unsigned short* __restrict__ B,
    const float* __restrict__ bias, int K, int N,
    unsigned short* __restrict__ outB, float* __restrict__ outF) {
  __shared__ unsigned short lA[64 * BGK_AS];
  __shared__ unsigned short lB[64 * BGK_BS];
  const int t = threadIdx.x;
  const int lane = t & 31;
  const int w = t >> 5, wm = w & 3, wn = w >> 2;
  const int m0 = blockIdx.y * 64;
  const int n0 = blockIdx.x * 128;

  const lds_u16* laBase = (const lds_u16*)(size_t)lds_off_of(&lA[0]);
  const lds_u16* lbBase = (const lds_u16*)(size_t)lds_off_of(&lB[0]);

  v8f acc[4];
#pragma unroll
  for (int j = 0; j < 4; ++j) acc[j] = {};

  for (int k0 = 0; k0 < K; k0 += 64) {
    __syncthreads();
#pragma unroll
    for (int p = 0; p < 2; ++p) {
      int idx = t + p * 256, r = idx >> 3, c = (idx & 7) * 8;
      *(v4u*)(lA + r * BGK_AS + c) = *(const v4u*)(A + (size_t)(m0 + r) * K + k0 + c);
    }
#pragma unroll
    for (int p = 0; p < 4; ++p) {
      int idx = t + p * 256, r = idx >> 4, c = (idx & 15) * 8;
      *(v4u*)(lB + r * BGK_BS + c) = *(const v4u*)(B + (size_t)(k0 + r) * N + n0 + c);
    }
    __syncthreads();
    FragBF af[2], bfr[2][4];
    const int arow = wm * 16 + (lane & 15);
    const int bkb = (lane >> 4) << 3;
#pragma unroll
    for (int ks = 0; ks < 2; ++ks) {
      const int ak = ks * 32 + ((lane >> 4) << 3);
      af[ks].u[0] = *(const lds_v4u*)(laBase + arow * BGK_AS + ak);
      af[ks].u[1] = *(const lds_v4u*)(laBase + arow * BGK_AS + ak + 16);
      const int bk = ks * 32 + bkb;
#pragma unroll
      for (int j = 0; j < 4; ++j) {
        const int bc = wn * 64 + j * 16 + (lane & 15);
#pragma unroll
        for (int s = 0; s < 16; ++s)
          ((unsigned short*)&bfr[ks][j])[s] = lbBase[(bk + (s < 8 ? s : s + 8)) * BGK_BS + bc];
      }
    }
#pragma unroll
    for (int ks = 0; ks < 2; ++ks)
#pragma unroll
      for (int j = 0; j < 4; ++j)
        acc[j] = __builtin_amdgcn_wmma_f32_16x16x32_bf16(
            false, af[ks].bf, false, bfr[ks][j].bf, (short)0, acc[j], false, false);
  }
#pragma unroll
  for (int j = 0; j < 4; ++j) {
    const int n = n0 + wn * 64 + j * 16 + (lane & 15);
    const float bv = bias[n];
    const int mrow = m0 + wm * 16 + ((lane >> 4) << 3);
#pragma unroll
    for (int r = 0; r < 8; ++r) {
      const float hv = fmaxf(acc[j][r] + bv, 0.f);
      const size_t o = (size_t)(mrow + r) * N + n;
      outB[o] = f2bf(hv);
      if (outF) outF[o] = hv;
    }
  }
}

// ---------------------------------------------------------------------------
// Kernel 4: elementwise f32 -> bf16
// ---------------------------------------------------------------------------
__global__ void cvt_bf16_kernel(const float* __restrict__ in,
                                unsigned short* __restrict__ out, int n) {
  int i = blockIdx.x * blockDim.x + threadIdx.x;
  if (i < n) out[i] = f2bf(in[i]);
}

// ---------------------------------------------------------------------------
// Kernel 5: greedy action per (t,n): argmax_a sum_k v7[k] * P[k, s0*4+a]
// (P rebuilt in f32).  One wave per row; also writes the "a" output.
// ---------------------------------------------------------------------------
__global__ __launch_bounds__(256) void next_idx_kernel(
    const float* __restrict__ val7, const float* __restrict__ T,
    const float* __restrict__ rowmax, const float* __restrict__ rowsum,
    const int* __restrict__ state_idx, int* __restrict__ nidx,
    float* __restrict__ a_out) {
  const int lane = threadIdx.x & 31;
  const int tn = blockIdx.x * 8 + (threadIdx.x >> 5);
  const int s0 = state_idx[tn];
  float a0 = 0.f, a1 = 0.f, a2 = 0.f, a3 = 0.f;
  for (int k = lane; k < 1024; k += 32) {
    const float vv = val7[(size_t)tn * 1024 + k];
    const float4 tv = *(const float4*)(T + (size_t)k * 4096 + s0 * 4);
    const float4 mv = *(const float4*)(rowmax + k * 4);
    const float4 sv = *(const float4*)(rowsum + k * 4);
    a0 += vv * __expf(tv.x - mv.x) / sv.x;
    a1 += vv * __expf(tv.y - mv.y) / sv.y;
    a2 += vv * __expf(tv.z - mv.z) / sv.z;
    a3 += vv * __expf(tv.w - mv.w) / sv.w;
  }
  for (int d = 1; d < 32; d <<= 1) {
    a0 += __shfl_xor(a0, d, 32); a1 += __shfl_xor(a1, d, 32);
    a2 += __shfl_xor(a2, d, 32); a3 += __shfl_xor(a3, d, 32);
  }
  if (lane == 0) {
    float best = a0; int bi = 0;
    if (a1 > best) { best = a1; bi = 1; }
    if (a2 > best) { best = a2; bi = 2; }
    if (a3 > best) { best = a3; bi = 3; }
    nidx[tn] = bi;
    a_out[tn * 2 + 0] = (float)(bi / 32 - s0 / 32);
    a_out[tn * 2 + 1] = (float)(bi % 32 - s0 % 32);
  }
}

// ---------------------------------------------------------------------------
// Kernel 6: x = concat(S[state_idx], S[next_idx])  (bf16 gather)
// ---------------------------------------------------------------------------
__global__ __launch_bounds__(256) void gather_x_kernel(
    const unsigned short* __restrict__ Sbf, const int* __restrict__ sidx,
    const int* __restrict__ nidx, unsigned short* __restrict__ x) {
  const int tn = blockIdx.x, t = threadIdx.x;
  const int s = sidx[tn], n = nidx[tn];
  x[(size_t)tn * 512 + t]       = Sbf[(size_t)s * 256 + t];
  x[(size_t)tn * 512 + 256 + t] = Sbf[(size_t)n * 256 + t];
}

// ---------------------------------------------------------------------------
// Kernel 7: actor/critic heads
// ---------------------------------------------------------------------------
__global__ __launch_bounds__(256) void heads_kernel(
    const float* __restrict__ h, const float* __restrict__ Wa,
    const float* __restrict__ ba, const float* __restrict__ Wc,
    const float* __restrict__ bc, float* __restrict__ a_probs,
    float* __restrict__ vout) {
  const int lane = threadIdx.x & 31;
  const int tn = blockIdx.x * 8 + (threadIdx.x >> 5);
  float l0 = 0.f, l1 = 0.f, l2 = 0.f, l3 = 0.f, lc = 0.f;
  for (int k = lane; k < 256; k += 32) {
    const float hv = h[(size_t)tn * 256 + k];
    const float4 wa = *(const float4*)(Wa + k * 4);
    l0 += hv * wa.x; l1 += hv * wa.y; l2 += hv * wa.z; l3 += hv * wa.w;
    lc += hv * Wc[k];
  }
  for (int d = 1; d < 32; d <<= 1) {
    l0 += __shfl_xor(l0, d, 32); l1 += __shfl_xor(l1, d, 32);
    l2 += __shfl_xor(l2, d, 32); l3 += __shfl_xor(l3, d, 32);
    lc += __shfl_xor(lc, d, 32);
  }
  if (lane == 0) {
    l0 += ba[0]; l1 += ba[1]; l2 += ba[2]; l3 += ba[3];
    const float mm = fmaxf(fmaxf(l0, l1), fmaxf(l2, l3));
    const float e0 = __expf(l0 - mm), e1 = __expf(l1 - mm);
    const float e2 = __expf(l2 - mm), e3 = __expf(l3 - mm);
    const float inv = 1.f / (e0 + e1 + e2 + e3);
    a_probs[tn * 4 + 0] = e0 * inv; a_probs[tn * 4 + 1] = e1 * inv;
    a_probs[tn * 4 + 2] = e2 * inv; a_probs[tn * 4 + 3] = e3 * inv;
    vout[tn] = lc + bc[0];
  }
}

// ---------------------------------------------------------------------------
extern "C" void kernel_launch(void* const* d_in, const int* in_sizes, int n_in,
                              void* d_out, int out_size, void* d_ws, size_t ws_size,
                              hipStream_t stream) {
  const float* rewards   = (const float*)d_in[0];   // [2048,1024]
  const int*   state_idx = (const int*)d_in[1];     // [2048]
  const float* S         = (const float*)d_in[2];   // [1024,256]
  const float* T_param   = (const float*)d_in[3];   // [1024,1024,4]
  const float* W1        = (const float*)d_in[4];   // [512,256]
  const float* b1        = (const float*)d_in[5];
  const float* W2        = (const float*)d_in[6];   // [256,256]
  const float* b2        = (const float*)d_in[7];
  const float* Wa        = (const float*)d_in[8];   // [256,4]
  const float* ba        = (const float*)d_in[9];
  const float* Wc        = (const float*)d_in[10];  // [256,1]
  const float* bc        = (const float*)d_in[11];
  float* out = (float*)d_out;
  (void)in_sizes; (void)n_in; (void)out_size; (void)ws_size;  // time_limit=8

  char* wp = (char*)d_ws;
  auto alloc = [&](size_t bytes) {
    char* p = wp; wp += (bytes + 255) & ~(size_t)255; return p;
  };
  unsigned short* P    = (unsigned short*)alloc(1024ull * 4096 * 2);
  unsigned short* vb0  = (unsigned short*)alloc(2048ull * 1024 * 2);
  unsigned short* vb1  = (unsigned short*)alloc(2048ull * 1024 * 2);
  float*          v7f  = (float*)alloc(2048ull * 1024 * 4);
  float*          rmax = (float*)alloc(4096 * 4);
  float*          rsum = (float*)alloc(4096 * 4);
  unsigned short* Sbf  = (unsigned short*)alloc(1024ull * 256 * 2);
  unsigned short* W1bf = (unsigned short*)alloc(512ull * 256 * 2);
  unsigned short* W2bf = (unsigned short*)alloc(256ull * 256 * 2);
  unsigned short* xbf  = (unsigned short*)alloc(2048ull * 512 * 2);
  unsigned short* h1bf = (unsigned short*)alloc(2048ull * 256 * 2);
  unsigned short* hbf  = (unsigned short*)alloc(2048ull * 256 * 2);
  int*            nidx = (int*)alloc(2048 * 4);

  float* out_a   = out;            // [2048,2]
  float* out_ap  = out + 4096;     // [2048,4]
  float* out_v   = out + 12288;    // [2048,1]
  float* out_h   = out + 14336;    // [2048,256]
  float* out_val = out + 538624;   // [2048,1024]

  softmax_p_kernel<<<1024, 256, 0, stream>>>(T_param, P, rmax, rsum);
  cvt_bf16_kernel<<<2097152 / 256, 256, 0, stream>>>(rewards, vb0, 2097152);
  cvt_bf16_kernel<<<262144 / 256, 256, 0, stream>>>(S, Sbf, 262144);
  cvt_bf16_kernel<<<131072 / 256, 256, 0, stream>>>(W1, W1bf, 131072);
  cvt_bf16_kernel<<<65536 / 256, 256, 0, stream>>>(W2, W2bf, 65536);

  unsigned short* cur = vb0;
  unsigned short* nxt = vb1;
  for (int it = 0; it < 6; ++it) {   // v2..v7
    bellman_gemm_kernel<<<dim3(32, 32), 256, 0, stream>>>(cur, P, rewards, nxt, v7f);
    unsigned short* tmp = cur; cur = nxt; nxt = tmp;
  }
  bellman_gemm_kernel<<<dim3(32, 32), 256, 0, stream>>>(cur, P, rewards, nxt, out_val);

  next_idx_kernel<<<256, 256, 0, stream>>>(v7f, T_param, rmax, rsum, state_idx, nidx, out_a);
  gather_x_kernel<<<2048, 256, 0, stream>>>(Sbf, state_idx, nidx, xbf);
  mlp_gemm_kernel<<<dim3(2, 32), 256, 0, stream>>>(xbf, W1bf, b1, 512, 256, h1bf, nullptr);
  mlp_gemm_kernel<<<dim3(2, 32), 256, 0, stream>>>(h1bf, W2bf, b2, 256, 256, hbf, out_h);
  heads_kernel<<<256, 256, 0, stream>>>(out_h, Wa, ba, Wc, bc, out_ap, out_v);
}
